// SRCNN_59820304499204
// MI455X (gfx1250) — 16-bit and 8-bit matrix operand order repaired and checked by reading, compiled, not run on hardware
//
#include <hip/hip_runtime.h>

typedef __attribute__((ext_vector_type(8)))  int      v8i;
typedef __attribute__((ext_vector_type(8)))  float    v8f;
typedef __attribute__((ext_vector_type(16))) _Float16 v16h;

#define IMG_HW  256
#define NPIX_F  1048576.0f

#define WS_SUM1 0
#define WS_SQ1  128
#define WS_A1   256
#define WS_C1   384
#define WS_SUM2 512
#define WS_SQ2  576
#define WS_A2   640
#define WS_C2   704
#define WS_TOT  768
#define WS_BP1  768
#define WS_BP2  2816
#define WS_BP3  4864

__device__ inline v8i wmma_i8(v8i a, v8i b, v8i c) {
  return __builtin_amdgcn_wmma_i32_16x16x64_iu8(true, a, true, b, c, false, false);
}
__device__ inline v8f wmma_h(v16h a, v16h b, v8f c) {
  return __builtin_amdgcn_wmma_f32_16x16x32_f16(false, a, false, b, (short)0, c,
                                                false, false);
}

__device__ inline v8i ldFrag8(const float* ws, int idx) {
  return *(const v8i*)((const int*)ws + idx);
}
__device__ inline v16h ldFragH(const float* ws, int idx) {
  return __builtin_bit_cast(v16h, *(const v8i*)((const int*)ws + idx));
}

template<int TH, int TW>
__device__ inline void loadXTile(signed char* xs, const float* __restrict__ x,
                                 int n, int oy, int ox) {
  const float* xi = x + n * (IMG_HW * IMG_HW);
  for (int i = threadIdx.x; i < TH * TW; i += blockDim.x) {
    int r = i / TW, c = i - r * TW;
    int gy = oy + r, gx = ox + c;
    signed char s = 0;
    if ((unsigned)gy < (unsigned)IMG_HW && (unsigned)gx < (unsigned)IMG_HW)
      s = (xi[gy * IMG_HW + gx] >= 0.0f) ? 1 : -1;
    xs[i] = s;
  }
}

template<int TW>
__device__ inline v8i buildA1(const signed char* xs, int rbase, int cbase, int lane) {
  const int m  = lane & 15;
  const int kh = (lane >> 4) << 3;
  v8i a;
#pragma unroll
  for (int v = 0; v < 8; ++v) {
    int w = 0;
#pragma unroll
    for (int b = 0; b < 4; ++b) {
      const int k = ((v >> 1) << 4) + kh + ((v & 1) << 2) + b;
      int s = 0;
      if (k < 49) {
        const int ky = k / 7;
        const int kx = k - ky * 7;
        s = (int)xs[(rbase + ky) * TW + (cbase + m + kx)];
      }
      w |= (s & 0xFF) << (b << 3);
    }
    a[v] = w;
  }
  return a;
}

__device__ inline v8i buildB1(const float* __restrict__ w1, int g, int lane) {
  const int ch = g * 16 + (lane & 15);
  const int kb = (lane >> 4) << 4;
  v8i r;
#pragma unroll
  for (int v = 0; v < 8; ++v) {
    int w = 0;
#pragma unroll
    for (int b = 0; b < 4; ++b) {
      const int k = ((v >> 1) << 4) + (kb >> 1) + ((v & 1) << 2) + b;
      int s = 0;
      if (k < 49) s = (w1[ch * 49 + k] >= 0.0f) ? 1 : -1;
      w |= (s & 0xFF) << (b << 3);
    }
    r[v] = w;
  }
  return r;
}

__device__ inline v8i buildB2(const float* __restrict__ w2, int half, int og, int lane) {
  const int oc = og * 16 + (lane & 15);
  const int kb = (lane >> 4) << 4;
  v8i r;
#pragma unroll
  for (int v = 0; v < 8; ++v) {
    int w = 0;
#pragma unroll
    for (int b = 0; b < 4; ++b) {
      const int ic = half * 64 + ((v >> 1) << 4) + (kb >> 1) + ((v & 1) << 2) + b;
      const int s = (w2[oc * 128 + ic] >= 0.0f) ? 1 : -1;
      w |= (s & 0xFF) << (b << 3);
    }
    r[v] = w;
  }
  return r;
}

__device__ inline v16h buildB3(const float* __restrict__ w3, int t, int lane,
                               int kpart) {
  const int j  = t * 16 + (lane & 15);
  const int kh = (lane >> 4) << 4;
  v16h b;
#pragma unroll
  for (int v = 0; v < 8; ++v) {
#pragma unroll
    for (int e = 0; e < 2; ++e) {
      const int c = kpart * 32 + kh + v * 2 + e;
      const float val = (j < 25) ? w3[c * 25 + j] : 0.0f;
      b[2 * v + e] = (_Float16)val;
    }
  }
  return b;
}

__device__ inline v8i buildA2(const int* s1w, int lane, int half) {
  const int m  = lane & 15;
  const int kh = (lane >> 4) << 3;
  v8i a;
#pragma unroll
  for (int v = 0; v < 8; ++v) {
    const int c = half * 64 + ((v >> 1) << 4) + kh + ((v & 1) << 2);
    a[v] = s1w[m * 32 + (c >> 2)];
  }
  return a;
}

__device__ inline void storeSigns(signed char* s1, const v8i d, int g, int lane,
                                  const float* a1s, const float* c1s) {
  const int n = g * 16 + (lane & 15);
  const float a = a1s[n], c = c1s[n];
  const int ph = (lane >> 4) << 3;
#pragma unroll
  for (int j = 0; j < 8; ++j) {
    const float v = (float)d[j] * a + c;
    s1[(ph + j) * 128 + n] = (v >= 0.0f) ? 1 : -1;
  }
}

__device__ inline v16h buildA3(const _Float16* act, int r, int cbase, int lane,
                               int kpart) {
  const int m  = lane & 15;
  const int kh = (lane >> 4) << 4;
  const _Float16* px = act + ((r * 48) + (cbase + m)) * 64 + kpart * 32;
  v16h a;
#pragma unroll
  for (int v = 0; v < 8; ++v) {
    const int k = kh + v * 2;
    a[2 * v]     = px[k];
    a[2 * v + 1] = px[k + 1];
  }
  return a;
}

__global__ void k_zero(float* ws) {
  if (threadIdx.x < WS_TOT) ws[threadIdx.x] = 0.0f;
}

__global__ void k_pack(const float* __restrict__ w1, const float* __restrict__ w2,
                       const float* __restrict__ w3, float* __restrict__ ws) {
  int* wi = (int*)ws;
  const int tid = threadIdx.x, lane = tid & 31;
  {
    const int g = tid >> 5;
    *(v8i*)&wi[WS_BP1 + tid * 8] = buildB1(w1, g, lane);
  }
  {
    const int half = tid >> 7, og = (tid >> 5) & 3;
    *(v8i*)&wi[WS_BP2 + tid * 8] = buildB2(w2, half, og, lane);
  }
  if (tid < 128) {
    const int kp = tid >> 6, t = (tid >> 5) & 1;
    v16h h = buildB3(w3, t, lane, kp);
    *(v8i*)&wi[WS_BP3 + tid * 8] = __builtin_bit_cast(v8i, h);
  }
}

__global__ __launch_bounds__(256) void k_conv1_stats(const float* __restrict__ x,
                                                     float* __restrict__ ws) {
  extern __shared__ char smem[];
  float* stat = (float*)smem;
  signed char* xs = (signed char*)(smem + 1024);
  const int tid = threadIdx.x, lane = tid & 31, wave = tid >> 5;
  const int bid = blockIdx.x;
  const int tx = bid & 15, ty = (bid >> 4) & 7, n = bid >> 7;
  const int x0 = tx * 16, y0 = ty * 32;

  if (tid < 256) stat[tid] = 0.0f;
  __builtin_prefetch(x + n * (IMG_HW * IMG_HW), 0, 0);
  loadXTile<38, 22>(xs, x, n, y0 - 3, x0 - 3);

  v8i b1f[8];
#pragma unroll
  for (int g = 0; g < 8; ++g) b1f[g] = ldFrag8(ws, WS_BP1 + (g * 32 + lane) * 8);
  __syncthreads();

  float sA[8], qA[8];
#pragma unroll
  for (int g = 0; g < 8; ++g) { sA[g] = 0.0f; qA[g] = 0.0f; }

  for (int i = 0; i < 4; ++i) {
    const int r = wave * 4 + i;
    const v8i a = buildA1<22>(xs, r, 0, lane);
#pragma unroll
    for (int g = 0; g < 8; ++g) {
      v8i d = {};
      d = wmma_i8(a, b1f[g], d);
#pragma unroll
      for (int j = 0; j < 8; ++j) {
        const float f = (float)d[j];
        sA[g] += f; qA[g] += f * f;
      }
    }
  }
  const int nch = lane & 15;
#pragma unroll
  for (int g = 0; g < 8; ++g) {
    atomicAdd(&stat[g * 16 + nch], sA[g]);
    atomicAdd(&stat[128 + g * 16 + nch], qA[g]);
  }
  __syncthreads();
  if (tid < 128)      atomicAdd(&ws[WS_SUM1 + tid], stat[tid]);
  else if (tid < 256) atomicAdd(&ws[WS_SQ1 + tid - 128], stat[tid]);
}

__global__ void k_bn1(const float* g1, const float* b1, float* ws) {
  const int c = threadIdx.x;
  if (c < 128) {
    const float invN = 1.0f / NPIX_F;
    const float mean = ws[WS_SUM1 + c] * invN;
    const float var  = ws[WS_SQ1 + c] * invN - mean * mean;
    const float inv  = g1[c] * rsqrtf(var + 1e-5f);
    ws[WS_A1 + c] = inv;
    ws[WS_C1 + c] = b1[c] - mean * inv;
  }
}

__global__ __launch_bounds__(256) void k_conv2_stats(const float* __restrict__ x,
                                                     float* __restrict__ ws) {
  extern __shared__ char smem[];
  float* a1s = (float*)smem;
  float* c1s = a1s + 128;
  float* stat = c1s + 128;
  int*  s1all = (int*)(smem + 384 * 4);
  signed char* xs = (signed char*)(s1all + 8 * 512);
  const int tid = threadIdx.x, lane = tid & 31, wave = tid >> 5;
  const int bid = blockIdx.x;
  const int tx = bid & 15, ty = (bid >> 4) & 7, n = bid >> 7;
  const int x0 = tx * 16, y0 = ty * 32;

  if (tid < 128) { a1s[tid] = ws[WS_A1 + tid]; c1s[tid] = ws[WS_C1 + tid];
                   stat[tid] = 0.0f; }
  loadXTile<38, 22>(xs, x, n, y0 - 3, x0 - 3);

  v8i b1f[8];
#pragma unroll
  for (int g = 0; g < 8; ++g) b1f[g] = ldFrag8(ws, WS_BP1 + (g * 32 + lane) * 8);
  v8i b2f[2][4];
#pragma unroll
  for (int h = 0; h < 2; ++h)
#pragma unroll
    for (int og = 0; og < 4; ++og)
      b2f[h][og] = ldFrag8(ws, WS_BP2 + ((h * 4 + og) * 32 + lane) * 8);
  __syncthreads();

  int* s1w = s1all + wave * 512;
  signed char* s1c = (signed char*)s1w;
  float sA[4] = {0, 0, 0, 0}, qA[4] = {0, 0, 0, 0};

  for (int i = 0; i < 4; ++i) {
    const int r = wave * 4 + i;
    const v8i a = buildA1<22>(xs, r, 0, lane);
#pragma unroll
    for (int g = 0; g < 8; ++g) {
      v8i d = {};
      d = wmma_i8(a, b1f[g], d);
      storeSigns(s1c, d, g, lane, a1s, c1s);
    }
    asm volatile("s_wait_dscnt 0" ::: "memory");
    const v8i a2l = buildA2(s1w, lane, 0);
    const v8i a2h = buildA2(s1w, lane, 1);
#pragma unroll
    for (int og = 0; og < 4; ++og) {
      v8i d2 = {};
      d2 = wmma_i8(a2l, b2f[0][og], d2);
      d2 = wmma_i8(a2h, b2f[1][og], d2);
#pragma unroll
      for (int j = 0; j < 8; ++j) {
        const float f = (float)d2[j];
        sA[og] += f; qA[og] += f * f;
      }
    }
  }
  const int nch = lane & 15;
#pragma unroll
  for (int og = 0; og < 4; ++og) {
    atomicAdd(&stat[og * 16 + nch], sA[og]);
    atomicAdd(&stat[64 + og * 16 + nch], qA[og]);
  }
  __syncthreads();
  if (tid < 64)       atomicAdd(&ws[WS_SUM2 + tid], stat[tid]);
  else if (tid < 128) atomicAdd(&ws[WS_SQ2 + tid - 64], stat[tid]);
}

__global__ void k_bn2(const float* g2, const float* b2, float* ws) {
  const int c = threadIdx.x;
  if (c < 64) {
    const float invN = 1.0f / NPIX_F;
    const float mean = ws[WS_SUM2 + c] * invN;
    const float var  = ws[WS_SQ2 + c] * invN - mean * mean;
    const float inv  = g2[c] * rsqrtf(var + 1e-5f);
    ws[WS_A2 + c] = inv;
    ws[WS_C2 + c] = b2[c] - mean * inv;
  }
}

__global__ __launch_bounds__(256) void k_final(const float* __restrict__ x,
                                               const float* __restrict__ ws,
                                               float* __restrict__ out) {
  extern __shared__ char smem[];
  float* a1s = (float*)smem;
  float* c1s = a1s + 128;
  float* a2s = c1s + 128;
  float* c2s = a2s + 64;
  _Float16* act2 = (_Float16*)(smem + 1536);
  _Float16* T    = act2 + 20 * 48 * 64;
  int* s1all = (int*)(smem + 1536 + (61440 + 24000) * 2);
  signed char* xs = (signed char*)(s1all + 8 * 512);
  float* xf = (float*)T;

  const int tid = threadIdx.x, lane = tid & 31, wave = tid >> 5;
  const int bid = blockIdx.x;
  const int tx = bid & 7, ty = (bid >> 3) & 15, n = bid >> 7;
  const int x0 = tx * 32, y0 = ty * 16;

  {
    const unsigned long long base =
        (unsigned long long)(uintptr_t)(x + n * (IMG_HW * IMG_HW));
    for (int i = tid; i < 26 * 54; i += 256) {
      const int r = i / 54, c = i - r * 54;
      int gy = y0 - 5 + r, gx = x0 - 5 + c;
      gy = gy < 0 ? 0 : (gy > 255 ? 255 : gy);
      gx = gx < 0 ? 0 : (gx > 255 ? 255 : gx);
      const int voff = (gy * IMG_HW + gx) * 4;
      const unsigned lds = (unsigned)(uintptr_t)(void*)&xf[i];
      asm volatile("global_load_async_to_lds_b32 %0, %1, %2"
                   :: "v"(lds), "v"(voff), "s"(base) : "memory");
    }
  }

  if (tid < 128)      { a1s[tid] = ws[WS_A1 + tid]; c1s[tid] = ws[WS_C1 + tid]; }
  else if (tid < 192) { a2s[tid - 128] = ws[WS_A2 + tid - 128];
                        c2s[tid - 128] = ws[WS_C2 + tid - 128]; }

  v8i b1f[8];
#pragma unroll
  for (int g = 0; g < 8; ++g) b1f[g] = ldFrag8(ws, WS_BP1 + (g * 32 + lane) * 8);
  v8i b2f[2][4];
#pragma unroll
  for (int h = 0; h < 2; ++h)
#pragma unroll
    for (int og = 0; og < 4; ++og)
      b2f[h][og] = ldFrag8(ws, WS_BP2 + ((h * 4 + og) * 32 + lane) * 8);
  v16h b3f[2][2];
#pragma unroll
  for (int kp = 0; kp < 2; ++kp)
#pragma unroll
    for (int t = 0; t < 2; ++t)
      b3f[kp][t] = ldFragH(ws, WS_BP3 + ((kp * 2 + t) * 32 + lane) * 8);

  asm volatile("s_wait_asynccnt 0" ::: "memory");
  __syncthreads();

  for (int i = tid; i < 26 * 54; i += 256) {
    const int r = i / 54, c = i - r * 54;
    const int gy = y0 - 5 + r, gx = x0 - 5 + c;
    signed char s = 0;
    if ((unsigned)gy < (unsigned)IMG_HW && (unsigned)gx < (unsigned)IMG_HW)
      s = (xf[i] >= 0.0f) ? 1 : -1;
    xs[i] = s;
  }
  __syncthreads();

  int* s1w = s1all + wave * 512;
  signed char* s1c = (signed char*)s1w;
  const int ph = (lane >> 4) << 3;

  for (int cell = wave; cell < 60; cell += 8) {
    const int r = cell / 3, cg = cell - r * 3;
    const v8i a = buildA1<54>(xs, r, cg * 16, lane);
#pragma unroll
    for (int g = 0; g < 8; ++g) {
      v8i d = {};
      d = wmma_i8(a, b1f[g], d);
      storeSigns(s1c, d, g, lane, a1s, c1s);
    }
    asm volatile("s_wait_dscnt 0" ::: "memory");
    const v8i a2l = buildA2(s1w, lane, 0);
    const v8i a2h = buildA2(s1w, lane, 1);
    const int gy = y0 - 2 + r;
#pragma unroll
    for (int og = 0; og < 4; ++og) {
      v8i d2 = {};
      d2 = wmma_i8(a2l, b2f[0][og], d2);
      d2 = wmma_i8(a2h, b2f[1][og], d2);
      const int n2 = og * 16 + (lane & 15);
      const float aa = a2s[n2], cc = c2s[n2];
#pragma unroll
      for (int j = 0; j < 8; ++j) {
        const int col = cg * 16 + ph + j;
        const int gx = x0 - 2 + col;
        const float v = (float)d2[j] * aa + cc;
        const float act = (v >= 0.0f) ? v : 0.5f * v;
        const bool in = ((unsigned)gy < (unsigned)IMG_HW) &&
                        ((unsigned)gx < (unsigned)IMG_HW);
        act2[(r * 48 + col) * 64 + n2] = in ? (_Float16)act : (_Float16)0.0f;
      }
    }
  }
  __syncthreads();

  for (int cell = wave; cell < 60; cell += 8) {
    const int r = cell / 3, cb = (cell - r * 3) * 16;
    const v16h a0 = buildA3(act2, r, cb, lane, 0);
    const v16h a1 = buildA3(act2, r, cb, lane, 1);
#pragma unroll
    for (int t = 0; t < 2; ++t) {
      v8f dd = {};
      dd = wmma_h(a0, b3f[0][t], dd);
      dd = wmma_h(a1, b3f[1][t], dd);
      const int j = t * 16 + (lane & 15);
      if (j < 25) {
#pragma unroll
        for (int jv = 0; jv < 8; ++jv) {
          const int m = ph + jv;
          T[(r * 48 + cb + m) * 25 + j] = (_Float16)dd[jv];
        }
      }
    }
  }
  __syncthreads();

#pragma unroll
  for (int p = 0; p < 2; ++p) {
    const int idx = tid + p * 256;
    const int oy = idx >> 5, ox = idx & 31;
    float acc = 0.0f;
#pragma unroll
    for (int ky = 0; ky < 5; ++ky)
#pragma unroll
      for (int kx = 0; kx < 5; ++kx)
        acc += (float)T[((oy + ky) * 48 + (ox + kx)) * 25 + ky * 5 + kx];
    out[n * (IMG_HW * IMG_HW) + (y0 + oy) * IMG_HW + (x0 + ox)] = acc;
  }
}

extern "C" void kernel_launch(void* const* d_in, const int* in_sizes, int n_in,
                              void* d_out, int out_size, void* d_ws, size_t ws_size,
                              hipStream_t stream) {
  (void)in_sizes; (void)n_in; (void)out_size; (void)ws_size;
  const float* x  = (const float*)d_in[0];
  const float* w1 = (const float*)d_in[1];
  const float* g1 = (const float*)d_in[2];
  const float* b1 = (const float*)d_in[3];
  const float* w2 = (const float*)d_in[4];
  const float* g2 = (const float*)d_in[5];
  const float* b2 = (const float*)d_in[6];
  const float* w3 = (const float*)d_in[7];
  float* out = (float*)d_out;
  float* ws  = (float*)d_ws;

  hipLaunchKernelGGL(k_zero, dim3(1), dim3(WS_TOT), 0, stream, ws);
  hipLaunchKernelGGL(k_pack, dim3(1), dim3(256), 0, stream, w1, w2, w3, ws);
  hipLaunchKernelGGL(k_conv1_stats, dim3(2048), dim3(256), 1888, stream, x, ws);
  hipLaunchKernelGGL(k_bn1, dim3(1), dim3(128), 0, stream, g1, b1, ws);
  hipLaunchKernelGGL(k_conv2_stats, dim3(2048), dim3(256), 18784, stream, x, ws);
  hipLaunchKernelGGL(k_bn2, dim3(1), dim3(64), 0, stream, g2, b2, ws);
  hipLaunchKernelGGL(k_final, dim3(2048), dim3(256), 190208, stream, x, ws, out);
}
